// MLSAForOneLevel_25812753449397
// MI455X (gfx1250) — matrix operand order repaired and checked by reading, not compiled, not run on hardware
//
#include <hip/hip_runtime.h>

typedef __attribute__((ext_vector_type(16))) __bf16 v16bf;
typedef __attribute__((ext_vector_type(8)))  __bf16 v8bf;
typedef __attribute__((ext_vector_type(8)))  float  v8f;

namespace {
constexpr int S_TOK = 4096;
constexpr int BATCH = 2;
constexpr int HEADS = 16;
constexpr int DH    = 128;
constexpr int DEG   = 4;                       // sparse degree
constexpr int SSP   = S_TOK / DEG;             // 1024 sparse tokens
constexpr int MTILE = 64;                      // query rows per workgroup
constexpr int NTILE = 32;                      // keys per inner block
constexpr int STRIDE_T = BATCH * HEADS * DH;   // 4096 floats
constexpr int STRIDE_B = HEADS * DH;           // 2048 floats
constexpr float SCALE  = 0.08838834764831845f; // 1/sqrt(128)
}

static __device__ __forceinline__ v16bf cat16(v8bf lo, v8bf hi) {
  return __builtin_shufflevector(lo, hi, 0,1,2,3,4,5,6,7,8,9,10,11,12,13,14,15);
}

static __device__ __forceinline__ v8bf pack8(float4 a, float4 b) {
  v8bf o;
  o[0]=(__bf16)a.x; o[1]=(__bf16)a.y; o[2]=(__bf16)a.z; o[3]=(__bf16)a.w;
  o[4]=(__bf16)b.x; o[5]=(__bf16)b.y; o[6]=(__bf16)b.z; o[7]=(__bf16)b.w;
  return o;
}

__global__ __launch_bounds__(128, 1)
void mlsa_flash_kernel(const float* __restrict__ q,
                       const float* __restrict__ k,
                       const float* __restrict__ v,
                       float* __restrict__ out,   // [S,B,H,Dh] fp32
                       float* __restrict__ den)   // [Ssp,B,H] fp32
{
  __shared__ __align__(16) unsigned char smem[16384 + 8192 + 8192 + 4096];
  __bf16* Qs  = (__bf16*)(smem);           // 64 x 128 bf16
  __bf16* Ks  = (__bf16*)(smem + 16384);   // 32 x 128 bf16, row-major
  __bf16* VTs = (__bf16*)(smem + 24576);   // 128 x 32 bf16, transposed
  __bf16* Ps  = (__bf16*)(smem + 32768);   // 4 waves x (16 x 32) bf16

  const int tid  = threadIdx.x;
  const int lane = tid & 31;
  const int wave = tid >> 5;
  const int hf   = lane >> 4;   // lane half (0/1), selects K sub-range
  const int l16  = lane & 15;

  const int tile = blockIdx.x & 15;        // 16 query tiles of 64 rows
  const int bh   = blockIdx.x >> 4;
  const int b    = bh >> 4;
  const int h    = bh & 15;
  const int r0   = h & (DEG - 1);          // per-head token offset
  const size_t base_bh = (size_t)b * STRIDE_B + (size_t)h * DH;

  // ---- stage gathered Q tile (64 x 128) into LDS as bf16 ----
  {
    const int row = tid >> 1;              // 0..63
    const int ch0 = (tid & 1) * 64;        // 0 or 64
    const int qi  = tile * MTILE + row;    // sparse row index
    const float4* gp = (const float4*)(q + (size_t)(qi*DEG + r0)*STRIDE_T + base_bh + ch0);
    __bf16* dst = Qs + row*DH + ch0;
    #pragma unroll
    for (int i = 0; i < 8; ++i)
      *(v8bf*)(dst + i*8) = pack8(gp[2*i], gp[2*i+1]);
  }
  __syncthreads();

  // ---- per-wave Q A-fragments (16 rows x 128 ch, 4 chunks of K=32) ----
  const int qrow0 = tile*MTILE + wave*16;
  v16bf qa[4];
  #pragma unroll
  for (int kc = 0; kc < 4; ++kc) {
    const __bf16* bp = Qs + (wave*16 + l16)*DH + kc*32;
    qa[kc] = cat16(*(const v8bf*)(bp + hf*16), *(const v8bf*)(bp + hf*16 + 8));
  }

  v8f O[8];                 // 16 x 128 fp32 context accumulator (8 C tiles)
  const v8f vzero = {};
  #pragma unroll
  for (int ct = 0; ct < 8; ++ct) O[ct] = vzero;
  float mst[8], lst[8];     // per-row running max / denom (row r or r+8 by half)
  #pragma unroll
  for (int r = 0; r < 8; ++r) { mst[r] = -1e30f; lst[r] = 0.0f; }

  const int nblocks = 2*tile + 2;          // causal bound: keys <= last row
  for (int jb = 0; jb < nblocks; ++jb) {
    __syncthreads();                       // protect Ks/VTs from prev readers
    // ---- cooperative K/V block load (32 gathered keys x 128 ch) ----
    {
      const int key = tid >> 2;            // 0..31
      const int ch0 = (tid & 3) * 32;      // 0,32,64,96
      const int jj  = jb*NTILE + key;      // sparse key index
      const size_t g = (size_t)(jj*DEG + r0)*STRIDE_T + base_bh + ch0;
      const float4* kp = (const float4*)(k + g);
      __bf16* kd = Ks + key*DH + ch0;
      #pragma unroll
      for (int i = 0; i < 4; ++i)
        *(v8bf*)(kd + i*8) = pack8(kp[2*i], kp[2*i+1]);
      const float4* vp = (const float4*)(v + g);
      #pragma unroll
      for (int i = 0; i < 8; ++i) {        // transpose V into [ch][key]
        float4 a = vp[i];
        VTs[(ch0 + 4*i + 0)*NTILE + key] = (__bf16)a.x;
        VTs[(ch0 + 4*i + 1)*NTILE + key] = (__bf16)a.y;
        VTs[(ch0 + 4*i + 2)*NTILE + key] = (__bf16)a.z;
        VTs[(ch0 + 4*i + 3)*NTILE + key] = (__bf16)a.w;
      }
    }
    __syncthreads();

    // ---- S = Q(16x128) * K^T(128x32): two 16x16 f32 tiles ----
    v8f s0 = vzero, s1 = vzero;
    #pragma unroll
    for (int kc = 0; kc < 4; ++kc) {
      const __bf16* b0p = Ks + l16*DH + kc*32 + hf*16;        // keys 0..15
      const __bf16* b1p = b0p + 16*DH;                        // keys 16..31
      v16bf b0 = cat16(*(const v8bf*)b0p, *(const v8bf*)(b0p + 8));
      v16bf b1 = cat16(*(const v8bf*)b1p, *(const v8bf*)(b1p + 8));
      s0 = __builtin_amdgcn_wmma_f32_16x16x32_bf16(false, qa[kc], false, b0, (short)0, s0, false, false);
      s1 = __builtin_amdgcn_wmma_f32_16x16x32_bf16(false, qa[kc], false, b1, (short)0, s1, false, false);
    }

    // ---- online softmax (row lives in 16 lanes of one half) ----
    const int nb = jb*NTILE;
    #pragma unroll
    for (int r = 0; r < 8; ++r) {
      const int mrow = qrow0 + r + hf*8;
      float e0 = s0[r]*SCALE;
      float e1 = s1[r]*SCALE;
      if (nb + l16      > mrow) e0 = -1e30f;   // causal mask
      if (nb + 16 + l16 > mrow) e1 = -1e30f;
      float rmax = fmaxf(e0, e1);
      rmax = fmaxf(rmax, __shfl_xor(rmax, 1, 32));
      rmax = fmaxf(rmax, __shfl_xor(rmax, 2, 32));
      rmax = fmaxf(rmax, __shfl_xor(rmax, 4, 32));
      rmax = fmaxf(rmax, __shfl_xor(rmax, 8, 32));
      const float mnew = fmaxf(mst[r], rmax);
      const float fac  = __expf(mst[r] - mnew);
      const float p0 = __expf(e0 - mnew);
      const float p1 = __expf(e1 - mnew);
      float rsum = p0 + p1;
      rsum += __shfl_xor(rsum, 1, 32);
      rsum += __shfl_xor(rsum, 2, 32);
      rsum += __shfl_xor(rsum, 4, 32);
      rsum += __shfl_xor(rsum, 8, 32);
      lst[r] = lst[r]*fac + rsum;
      mst[r] = mnew;
      #pragma unroll
      for (int ct = 0; ct < 8; ++ct) O[ct][r] *= fac;         // rescale O
      __bf16* pw = Ps + wave*512 + (r + hf*8)*32;             // stage P
      pw[l16]      = (__bf16)p0;
      pw[16 + l16] = (__bf16)p1;
    }

    // ---- O += P(16x32) * V(32x128) ----
    {
      const __bf16* pp = Ps + wave*512 + l16*32;
      v16bf pa = cat16(*(const v8bf*)(pp + hf*16), *(const v8bf*)(pp + hf*16 + 8));
      #pragma unroll
      for (int ct = 0; ct < 8; ++ct) {
        const __bf16* vb = VTs + (ct*16 + l16)*NTILE + hf*16;
        v16bf bv = cat16(*(const v8bf*)vb, *(const v8bf*)(vb + 8));
        O[ct] = __builtin_amdgcn_wmma_f32_16x16x32_bf16(false, pa, false, bv, (short)0, O[ct], false, false);
      }
    }
  }

  // ---- epilogue: normalize, scatter ctx rows, zero sibling rows, denom ----
  float invl[8];
  #pragma unroll
  for (int r = 0; r < 8; ++r) invl[r] = 1.0f / lst[r];

  #pragma unroll
  for (int ct = 0; ct < 8; ++ct) {
    #pragma unroll
    for (int r = 0; r < 8; ++r) {
      const int m = qrow0 + r + hf*8;
      out[(size_t)(m*DEG + r0)*STRIDE_T + base_bh + ct*16 + l16] = O[ct][r]*invl[r];
    }
  }

  const float4 z4 = make_float4(0.f, 0.f, 0.f, 0.f);
  #pragma unroll
  for (int rr = 0; rr < 16; ++rr) {
    const int m = qrow0 + rr;
    #pragma unroll
    for (int z = 1; z < 4; ++z) {          // tokens this head does not select
      const int tg = m*DEG + ((r0 + z) & 3);
      *(float4*)(out + (size_t)tg*STRIDE_T + base_bh + lane*4) = z4;
    }
  }

  if (l16 == 0) {                          // lanes 0 and 16 cover rows 0..7 / 8..15
    #pragma unroll
    for (int r = 0; r < 8; ++r) {
      const int m = qrow0 + r + hf*8;
      den[(size_t)m*(BATCH*HEADS) + b*HEADS + h] = lst[r];
    }
  }
}

extern "C" void kernel_launch(void* const* d_in, const int* in_sizes, int n_in,
                              void* d_out, int out_size, void* d_ws, size_t ws_size,
                              hipStream_t stream) {
  (void)in_sizes; (void)n_in; (void)d_ws; (void)ws_size; (void)out_size;
  const float* q = (const float*)d_in[0];
  const float* k = (const float*)d_in[1];
  const float* v = (const float*)d_in[2];
  float* out = (float*)d_out;
  float* den = out + (size_t)S_TOK * BATCH * HEADS * DH;   // denom follows ctx
  dim3 grid(BATCH * HEADS * (SSP / MTILE));                // 512 workgroups
  dim3 block(128);                                         // 4 wave32s
  hipLaunchKernelGGL(mlsa_flash_kernel, grid, block, 0, stream, q, k, v, out, den);
}
